// GraphAttentionEncoder_32487132627370
// MI455X (gfx1250) — matrix operand order repaired and checked by reading, not compiled, not run on hardware
//
#include <hip/hip_runtime.h>
#include <hip/hip_bf16.h>

// ---------------------------------------------------------------------------
// Problem constants (from reference)
// ---------------------------------------------------------------------------
constexpr int BATCH = 8;
constexpr int NNODE = 512;
constexpr int INF0  = 128;
constexpr int OUTF  = 256;
constexpr int EF    = 16;
constexpr int NLAY  = 3;
constexpr int POOLH = 128;

typedef __attribute__((ext_vector_type(16))) _Float16 hv16;
typedef __attribute__((ext_vector_type(4)))  _Float16 hv4;
typedef __attribute__((ext_vector_type(8)))  float    fv8;

// ---------------------------------------------------------------------------
// device helpers
// ---------------------------------------------------------------------------
__device__ __forceinline__ float lrelu_f(float x) { return x >= 0.f ? x : 0.2f * x; }
__device__ __forceinline__ float gelu_f(float x) {
    float x3 = x * x * x;
    return 0.5f * x * (1.f + tanhf(0.7978845608028654f * (x + 0.044715f * x3)));
}
__device__ __forceinline__ float celu_f(float x) { return x > 0.f ? x : expm1f(x); }
__device__ __forceinline__ float act_f(float x, int a) {
    if (a == 1) return lrelu_f(x);
    if (a == 2) return gelu_f(x);
    if (a == 3) return celu_f(x);
    return x;
}

// A-fragment: 16x32 f16, per ISA table (lanes 0-15: K 0-7 / 16-23,
// lanes 16-31: K 8-15 / 24-31). smem row-major [16][stride].
__device__ __forceinline__ hv16 frag_a(const _Float16* s, int stride, int lane) {
    int row = lane & 15;
    int kb  = (lane & 16) ? 8 : 0;
    const _Float16* r = s + row * stride;
    hv16 f;
#pragma unroll
    for (int e = 0; e < 16; ++e) {
        int k = kb + (e < 8 ? e : e + 8);
        f[e] = r[k];
    }
    return f;
}

// B-fragment: 32x16 f16. lane = column; lanes 0-15 hold K 0-7 / 16-23,
// lanes 16-31 hold K 8-15 / 24-31. smem stored n-major [16][stride(k)].
__device__ __forceinline__ hv16 frag_b(const _Float16* s, int stride, int lane) {
    int n  = lane & 15;
    int kb = (lane & 16) ? 8 : 0;
    const _Float16* r = s + n * stride;
    hv16 f;
#pragma unroll
    for (int e = 0; e < 16; ++e) f[e] = r[kb + (e < 8 ? e : e + 8)];
    return f;
}

__device__ __forceinline__ hv4 cvt4(float4 v) {
    hv4 h;
    h[0] = (_Float16)v.x; h[1] = (_Float16)v.y;
    h[2] = (_Float16)v.z; h[3] = (_Float16)v.w;
    return h;
}

// ---------------------------------------------------------------------------
// Generic batched WMMA GEMM:  C = act(alpha * A @ op(B) + bias + R)
//   A: [M,K] row-major (lda), per-batch offsets b*sAb + h*sAh (z = b*H + h)
//   B: kmajor=0 -> "weight" layout  B[n][k] (ldb over n)
//      kmajor=1 -> "value"  layout  B[k][n] (ldb over k)
//   block: 128 threads (4 waves), tile 64(M) x NT(N), K stepped by 32.
//   Register-staged software pipeline: next K-tile's float4 loads are issued
//   while the current tile's WMMAs execute; f32->f16 conversion at LDS store.
// ---------------------------------------------------------------------------
struct GemmP {
    const float* A; int lda; long sAb, sAh;
    const float* B; int ldb; long sBb, sBh; int kmajor;
    const float* bias;
    const float* R; int ldr; long sRb, sRh;
    float* C; int ldc; long sCb, sCh;
    int M, Nc, K, H;
    float alpha; int act;
};

template <int NT>
__global__ __launch_bounds__(128) void gemm_wmma_kernel(GemmP p) {
    constexpr int NSUB = NT / 16;   // 16-wide N subtiles per block tile
    constexpr int BQ   = NT / 16;   // float4 passes for the B tile
    __shared__ _Float16 As[64][40];
    __shared__ _Float16 Bs[NT][40];

    const int z  = blockIdx.z;
    const int bb = z / p.H;
    const int hh = z % p.H;
    const float* A = p.A + (long)bb * p.sAb + (long)hh * p.sAh;
    const float* B = p.B + (long)bb * p.sBb + (long)hh * p.sBh;
    const float* R = p.R ? p.R + (long)bb * p.sRb + (long)hh * p.sRh : nullptr;
    float* C = p.C + (long)bb * p.sCb + (long)hh * p.sCh;

    const int m0   = blockIdx.x * 64;
    const int n0   = blockIdx.y * NT;
    const int tid  = threadIdx.x;
    const int wave = tid >> 5;
    const int lane = tid & 31;

    // ---- loop-invariant per-thread quad coordinates & base pointers ----
    int am[4], ac4[4];
    const float* aptr[4];
#pragma unroll
    for (int it = 0; it < 4; ++it) {
        int q   = tid + it * 128;        // 512 quads of the 64x32 A tile
        am[it]  = q >> 3;
        ac4[it] = (q & 7) * 4;
        aptr[it] = A + (long)(m0 + am[it]) * p.lda + ac4[it];
    }
    int bk[BQ], bn4[BQ];
    const float* bptr[BQ];
    if (p.kmajor) {
#pragma unroll
        for (int it = 0; it < BQ; ++it) {
            int q   = tid + it * 128;    // 8*NT quads of the 32xNT tile
            bk[it]  = q / (NT / 4);
            bn4[it] = (q % (NT / 4)) * 4;
            bptr[it] = B + (long)bk[it] * p.ldb + n0 + bn4[it];
        }
    } else {
#pragma unroll
        for (int it = 0; it < BQ; ++it) {
            int q   = tid + it * 128;    // NT*8 quads of the NTx32 tile
            bk[it]  = q >> 3;            // n index
            bn4[it] = (q & 7) * 4;       // k quad
            bptr[it] = B + (long)(n0 + bk[it]) * p.ldb + bn4[it];
        }
    }

    fv8 acc[NSUB];
#pragma unroll
    for (int s = 0; s < NSUB; ++s) acc[s] = fv8{};

    // ---- prologue: issue all loads for k0 = 0 ----
    float4 ra[4], rb[BQ];
#pragma unroll
    for (int it = 0; it < 4; ++it) ra[it] = *(const float4*)(aptr[it]);
    if (p.kmajor) {
#pragma unroll
        for (int it = 0; it < BQ; ++it) rb[it] = *(const float4*)(bptr[it]);
    } else {
#pragma unroll
        for (int it = 0; it < BQ; ++it) rb[it] = *(const float4*)(bptr[it]);
    }

    for (int k0 = 0; k0 < p.K; k0 += 32) {
        // ---- stage registers -> LDS (f32 -> f16) ----
#pragma unroll
        for (int it = 0; it < 4; ++it)
            *(hv4*)&As[am[it]][ac4[it]] = cvt4(ra[it]);
        if (p.kmajor) {
#pragma unroll
            for (int it = 0; it < BQ; ++it) {
                Bs[bn4[it] + 0][bk[it]] = (_Float16)rb[it].x;
                Bs[bn4[it] + 1][bk[it]] = (_Float16)rb[it].y;
                Bs[bn4[it] + 2][bk[it]] = (_Float16)rb[it].z;
                Bs[bn4[it] + 3][bk[it]] = (_Float16)rb[it].w;
            }
        } else {
#pragma unroll
            for (int it = 0; it < BQ; ++it)
                *(hv4*)&Bs[bk[it]][bn4[it]] = cvt4(rb[it]);
        }
        __syncthreads();

        // ---- issue next tile's loads (hidden under the WMMAs below) ----
        const int k1 = k0 + 32;
        if (k1 < p.K) {
#pragma unroll
            for (int it = 0; it < 4; ++it)
                ra[it] = *(const float4*)(aptr[it] + k1);
            if (p.kmajor) {
#pragma unroll
                for (int it = 0; it < BQ; ++it)
                    rb[it] = *(const float4*)(bptr[it] + (long)k1 * p.ldb);
            } else {
#pragma unroll
                for (int it = 0; it < BQ; ++it)
                    rb[it] = *(const float4*)(bptr[it] + k1);
            }
        }

        // ---- WMMA on current tile ----
        hv16 fa = frag_a(&As[wave * 16][0], 40, lane);
#pragma unroll
        for (int s = 0; s < NSUB; ++s) {
            hv16 fb = frag_b(&Bs[s * 16][0], 40, lane);
            acc[s] = __builtin_amdgcn_wmma_f32_16x16x32_f16(false, fa, false, fb,
                                                            (short)0, acc[s], false, false);
        }
        __syncthreads();
    }

    // epilogue — C/D layout: VGPR i: lanes 0-15 -> M=i, lanes 16-31 -> M=i+8
    const int nlocal = lane & 15;
    const int mbase  = m0 + wave * 16 + ((lane >> 4) << 3);
#pragma unroll
    for (int s = 0; s < NSUB; ++s) {
        int n = n0 + s * 16 + nlocal;
        float bia = p.bias ? p.bias[n] : 0.f;
#pragma unroll
        for (int i = 0; i < 8; ++i) {
            int m = mbase + i;
            float v = acc[s][i] * p.alpha + bia;
            if (R) v += R[(long)m * p.ldr + n];
            C[(long)m * p.ldc + n] = act_f(v, p.act);
        }
    }
}

// ---------------------------------------------------------------------------
// Row LayerNorm: one block (256 thr) per row; optional lrelu on input.
// ---------------------------------------------------------------------------
__global__ __launch_bounds__(256) void ln_kernel(const float* __restrict__ in,
                                                 const float* __restrict__ g,
                                                 const float* __restrict__ b,
                                                 float* __restrict__ out,
                                                 int D, int act) {
    __shared__ float red[256];
    const long row = blockIdx.x;
    const float* x = in + row * D;
    float* o = out + row * D;
    const int t = threadIdx.x;

    float s = 0.f;
    for (int d = t; d < D; d += 256) {
        float v = x[d]; if (act == 1) v = lrelu_f(v); s += v;
    }
    red[t] = s; __syncthreads();
    for (int k = 128; k > 0; k >>= 1) { if (t < k) red[t] += red[t + k]; __syncthreads(); }
    float mean = red[0] / D; __syncthreads();

    s = 0.f;
    for (int d = t; d < D; d += 256) {
        float v = x[d]; if (act == 1) v = lrelu_f(v); v -= mean; s += v * v;
    }
    red[t] = s; __syncthreads();
    for (int k = 128; k > 0; k >>= 1) { if (t < k) red[t] += red[t + k]; __syncthreads(); }
    float rstd = rsqrtf(red[0] / D + 1e-5f); __syncthreads();

    for (int d = t; d < D; d += 256) {
        float v = x[d]; if (act == 1) v = lrelu_f(v);
        o[d] = (v - mean) * rstd * g[d] + b[d];
    }
}

// ---------------------------------------------------------------------------
// a_sum[d] = sum_h attn[h][d]
// ---------------------------------------------------------------------------
__global__ void asum_kernel(const float* __restrict__ attn, float* __restrict__ a_sum,
                            int H, int D) {
    int d = blockIdx.x * blockDim.x + threadIdx.x;
    if (d < D) {
        float s = 0.f;
        for (int h = 0; h < H; ++h) s += attn[h * D + d];
        a_sum[d] = s;
    }
}

// ---------------------------------------------------------------------------
// Fused edge pipeline (per edge, in-place capable):
//   n = LN(e); t = gelu(n@W1^T+b1); o = t@W2^T+b2 + e;  eOut=o;  logitE=dot(o,a3)
// ---------------------------------------------------------------------------
__global__ __launch_bounds__(256) void edge_kernel(const float* __restrict__ eIn,
                                                   float* __restrict__ eOut,
                                                   float* __restrict__ logitE,
                                                   const float* ln2g, const float* ln2b,
                                                   const float* w1, const float* b1,
                                                   const float* w2, const float* b2,
                                                   const float* a3) {
    __shared__ float sw1[32 * 16], sw2[16 * 32];
    __shared__ float sb1[32], sb2[16], sg[16], sbt[16], sa3[16];
    const int t = threadIdx.x;
    for (int i = t; i < 512; i += 256) { sw1[i] = w1[i]; sw2[i] = w2[i]; }
    if (t < 32) sb1[t] = b1[t];
    if (t < 16) { sb2[t] = b2[t]; sg[t] = ln2g[t]; sbt[t] = ln2b[t]; sa3[t] = a3[t]; }
    __syncthreads();

    const long gid = (long)blockIdx.x * 256 + t;   // one edge per thread
    const float* ep = eIn + gid * EF;

    float e[EF];
    float4 v0 = *(const float4*)(ep + 0);
    float4 v1 = *(const float4*)(ep + 4);
    float4 v2 = *(const float4*)(ep + 8);
    float4 v3 = *(const float4*)(ep + 12);
    e[0]=v0.x; e[1]=v0.y; e[2]=v0.z; e[3]=v0.w;
    e[4]=v1.x; e[5]=v1.y; e[6]=v1.z; e[7]=v1.w;
    e[8]=v2.x; e[9]=v2.y; e[10]=v2.z; e[11]=v2.w;
    e[12]=v3.x; e[13]=v3.y; e[14]=v3.z; e[15]=v3.w;

    float mean = 0.f;
#pragma unroll
    for (int k = 0; k < EF; ++k) mean += e[k];
    mean *= (1.f / EF);
    float var = 0.f;
#pragma unroll
    for (int k = 0; k < EF; ++k) { float d = e[k] - mean; var += d * d; }
    var *= (1.f / EF);
    float rstd = rsqrtf(var + 1e-5f);

    float nn[EF];
#pragma unroll
    for (int k = 0; k < EF; ++k) nn[k] = (e[k] - mean) * rstd * sg[k] + sbt[k];

    float o[EF];
#pragma unroll
    for (int k = 0; k < EF; ++k) o[k] = sb2[k] + e[k];   // bias + residual

    for (int j = 0; j < 32; ++j) {
        float a = sb1[j];
#pragma unroll
        for (int k = 0; k < EF; ++k) a += nn[k] * sw1[j * 16 + k];
        a = gelu_f(a);
#pragma unroll
        for (int k = 0; k < EF; ++k) o[k] += a * sw2[k * 32 + j];
    }

    float lg = 0.f;
#pragma unroll
    for (int k = 0; k < EF; ++k) lg += o[k] * sa3[k];

    float* op = eOut + gid * EF;
    float4 s0 = { o[0], o[1], o[2], o[3] };
    float4 s1 = { o[4], o[5], o[6], o[7] };
    float4 s2 = { o[8], o[9], o[10], o[11] };
    float4 s3 = { o[12], o[13], o[14], o[15] };
    *(float4*)(op + 0)  = s0;
    *(float4*)(op + 4)  = s1;
    *(float4*)(op + 8)  = s2;
    *(float4*)(op + 12) = s3;
    logitE[gid] = lg;
}

// ---------------------------------------------------------------------------
// logit_i / logit_j: li[(b*H+h)*N + n] = dot(h[b,n,h,:], a1); lj likewise a2
// ---------------------------------------------------------------------------
__global__ void logit_kernel(const float* __restrict__ hbuf, const float* __restrict__ a_sum,
                             float* __restrict__ li, float* __restrict__ lj,
                             int H, int hs) {
    long g = (long)blockIdx.x * blockDim.x + threadIdx.x;
    long tot = (long)BATCH * NNODE * H;
    if (g >= tot) return;
    int b = (int)(g / ((long)NNODE * H));
    int rem = (int)(g % ((long)NNODE * H));
    int n = rem / H;
    int h = rem % H;
    const float* x = hbuf + ((long)b * NNODE + n) * OUTF + h * hs;
    float s1 = 0.f, s2 = 0.f;
    for (int d = 0; d < hs; ++d) {
        float v = x[d];
        s1 += v * a_sum[d];
        s2 += v * a_sum[hs + d];
    }
    long oi = ((long)b * H + h) * NNODE + n;
    li[oi] = s1;
    lj[oi] = s2;
}

// ---------------------------------------------------------------------------
// GAT coefficients: one wave32 per (b,h,i) row of 512.
// val = li + lj + logitE; mask -> NEG_INF; lrelu; softmax over j.
// ---------------------------------------------------------------------------
__global__ __launch_bounds__(256) void coeffs_kernel(const float* __restrict__ li,
                                                     const float* __restrict__ lj,
                                                     const float* __restrict__ lE,
                                                     const int* __restrict__ adj,
                                                     float* __restrict__ out, int H) {
    const long wid  = ((long)blockIdx.x * blockDim.x + threadIdx.x) >> 5;
    const int lane  = threadIdx.x & 31;
    const int b   = (int)(wid / ((long)H * NNODE));
    const int rem = (int)(wid % ((long)H * NNODE));
    const int h   = rem / NNODE;
    const int i   = rem % NNODE;

    const float lii = li[((long)b * H + h) * NNODE + i];
    const float* ljr = lj + ((long)b * H + h) * NNODE;
    const float* ler = lE + ((long)b * NNODE + i) * NNODE;
    const int*   ar  = adj + ((long)b * NNODE + i) * NNODE;

    float v[16];
    float mx = -3.4e38f;
#pragma unroll
    for (int c = 0; c < 16; ++c) {
        int j = lane + c * 32;
        float val = lii + ljr[j] + ler[j];
        if (ar[j] == 0) val = -1e30f;
        val = lrelu_f(val);
        v[c] = val;
        mx = fmaxf(mx, val);
    }
    for (int o = 16; o > 0; o >>= 1) mx = fmaxf(mx, __shfl_xor(mx, o, 32));
    float sum = 0.f;
#pragma unroll
    for (int c = 0; c < 16; ++c) { v[c] = __expf(v[c] - mx); sum += v[c]; }
    for (int o = 16; o > 0; o >>= 1) sum += __shfl_xor(sum, o, 32);
    float inv = (sum > 0.f) ? 1.f / sum : 0.f;
    float* orow = out + wid * NNODE;
#pragma unroll
    for (int c = 0; c < 16; ++c) orow[lane + c * 32] = v[c] * inv;
}

// ---------------------------------------------------------------------------
// In-place softmax over 512-length rows (one wave per row)
// ---------------------------------------------------------------------------
__global__ __launch_bounds__(256) void softmax512_kernel(float* __restrict__ buf) {
    const long wid = ((long)blockIdx.x * blockDim.x + threadIdx.x) >> 5;
    const int lane = threadIdx.x & 31;
    float* row = buf + wid * NNODE;
    float v[16];
    float mx = -3.4e38f;
#pragma unroll
    for (int c = 0; c < 16; ++c) { v[c] = row[lane + c * 32]; mx = fmaxf(mx, v[c]); }
    for (int o = 16; o > 0; o >>= 1) mx = fmaxf(mx, __shfl_xor(mx, o, 32));
    float sum = 0.f;
#pragma unroll
    for (int c = 0; c < 16; ++c) { v[c] = __expf(v[c] - mx); sum += v[c]; }
    for (int o = 16; o > 0; o >>= 1) sum += __shfl_xor(sum, o, 32);
    float inv = 1.f / sum;
#pragma unroll
    for (int c = 0; c < 16; ++c) row[lane + c * 32] = v[c] * inv;
}

// ---------------------------------------------------------------------------
// fused = res_p + x2 + gamma * g
// ---------------------------------------------------------------------------
__global__ void add3_kernel(const float* __restrict__ a, const float* __restrict__ b,
                            const float* __restrict__ c, const float* __restrict__ gamma,
                            float* __restrict__ out, long n) {
    long i = (long)blockIdx.x * blockDim.x + threadIdx.x;
    if (i < n) out[i] = a[i] + b[i] + gamma[0] * c[i];
}

// ---------------------------------------------------------------------------
// Pooling stage 1: attention-pool over nodes -> pooled[b][256]
// ---------------------------------------------------------------------------
__global__ __launch_bounds__(256) void pool1_kernel(const float* __restrict__ x,
                                                    const float* __restrict__ paw,
                                                    const float* __restrict__ pab,
                                                    float* __restrict__ pooled) {
    __shared__ float sc[NNODE];
    __shared__ float red[256];
    const int b = blockIdx.x;
    const int t = threadIdx.x;
    const float* xb = x + (long)b * NNODE * OUTF;

    for (int n = t; n < NNODE; n += 256) {
        float s = pab[0];
        const float* r = xb + (long)n * OUTF;
        for (int d = 0; d < OUTF; ++d) s += r[d] * paw[d];
        sc[n] = s;
    }
    __syncthreads();
    float mx = -3.4e38f;
    for (int n = t; n < NNODE; n += 256) mx = fmaxf(mx, sc[n]);
    red[t] = mx; __syncthreads();
    for (int k = 128; k > 0; k >>= 1) { if (t < k) red[t] = fmaxf(red[t], red[t + k]); __syncthreads(); }
    mx = red[0]; __syncthreads();
    float sum = 0.f;
    for (int n = t; n < NNODE; n += 256) { float e = __expf(sc[n] - mx); sc[n] = e; sum += e; }
    red[t] = sum; __syncthreads();
    for (int k = 128; k > 0; k >>= 1) { if (t < k) red[t] += red[t + k]; __syncthreads(); }
    float inv = 1.f / red[0]; __syncthreads();

    for (int d = t; d < OUTF; d += 256) {
        float acc = 0.f;
        for (int n = 0; n < NNODE; ++n) acc += sc[n] * xb[(long)n * OUTF + d];
        pooled[(long)b * OUTF + d] = acc * inv;
    }
}

// ---------------------------------------------------------------------------
// Pooling stage 2: out[b] = w2 . gelu(LN(pooled @ w1^T + b1)) + b2
// ---------------------------------------------------------------------------
__global__ __launch_bounds__(POOLH) void pool2_kernel(const float* __restrict__ pooled,
                                                      const float* __restrict__ w1,
                                                      const float* __restrict__ b1,
                                                      const float* __restrict__ lg,
                                                      const float* __restrict__ lb,
                                                      const float* __restrict__ w2,
                                                      const float* __restrict__ b2,
                                                      float* __restrict__ out) {
    __shared__ float red[POOLH];
    const int b = blockIdx.x;
    const int t = threadIdx.x;
    const float* pb = pooled + (long)b * OUTF;

    float v = b1[t];
    for (int d = 0; d < OUTF; ++d) v += pb[d] * w1[t * OUTF + d];

    red[t] = v; __syncthreads();
    for (int k = POOLH / 2; k > 0; k >>= 1) { if (t < k) red[t] += red[t + k]; __syncthreads(); }
    float mean = red[0] / POOLH; __syncthreads();
    float c = v - mean;
    red[t] = c * c; __syncthreads();
    for (int k = POOLH / 2; k > 0; k >>= 1) { if (t < k) red[t] += red[t + k]; __syncthreads(); }
    float rstd = rsqrtf(red[0] / POOLH + 1e-5f); __syncthreads();
    float y = gelu_f(c * rstd * lg[t] + lb[t]);
    red[t] = y * w2[t]; __syncthreads();
    for (int k = POOLH / 2; k > 0; k >>= 1) { if (t < k) red[t] += red[t + k]; __syncthreads(); }
    if (t == 0) out[b] = red[0] + b2[0];
}

// ---------------------------------------------------------------------------
// Host-side launch helper for the generic GEMM (picks 64- or 32-wide N tile)
// ---------------------------------------------------------------------------
static void gemm(hipStream_t s,
                 const float* A, int lda, long sAb, long sAh,
                 const float* B, int ldb, long sBb, long sBh, int kmajor,
                 const float* bias,
                 const float* R, int ldr, long sRb, long sRh,
                 float* C, int ldc, long sCb, long sCh,
                 int M, int Nc, int K, int nb, int H, float alpha, int act) {
    GemmP p;
    p.A = A; p.lda = lda; p.sAb = sAb; p.sAh = sAh;
    p.B = B; p.ldb = ldb; p.sBb = sBb; p.sBh = sBh; p.kmajor = kmajor;
    p.bias = bias;
    p.R = R; p.ldr = ldr; p.sRb = sRb; p.sRh = sRh;
    p.C = C; p.ldc = ldc; p.sCb = sCb; p.sCh = sCh;
    p.M = M; p.Nc = Nc; p.K = K; p.H = H; p.alpha = alpha; p.act = act;
    if (Nc % 64 == 0) {
        dim3 g(M / 64, Nc / 64, nb);
        gemm_wmma_kernel<64><<<g, 128, 0, s>>>(p);
    } else {
        dim3 g(M / 64, Nc / 32, nb);
        gemm_wmma_kernel<32><<<g, 128, 0, s>>>(p);
    }
}

// simple (non-batched, weight-layout B) wrapper
static void gemm_w(hipStream_t s, const float* A, int lda, const float* W, int K_ld,
                   const float* bias, const float* R, int ldr,
                   float* C, int ldc, int M, int Nc, int K, int act) {
    gemm(s, A, lda, 0, 0, W, K_ld, 0, 0, 0, bias, R, ldr, 0, 0,
         C, ldc, 0, 0, M, Nc, K, 1, 1, 1.f, act);
}

struct LayerPtrs {
    const float *ln1_g, *ln1_b, *np_w, *np_b, *ln2_g, *ln2_b;
    const float *em_w1, *em_b1, *em_w2, *em_b2, *attn;
    const float *op_w, *op_b, *lno_g, *lno_b;
    const float *ga_in_w, *ga_in_b, *ga_out_w, *ga_out_b;
    const float *gamma, *n_g, *n_b, *mlp_w1, *mlp_b1, *mlp_w2, *mlp_b2;
    const float *rp_w, *rp_b;
};

extern "C" void kernel_launch(void* const* d_in, const int* in_sizes, int n_in,
                              void* d_out, int out_size, void* d_ws, size_t ws_size,
                              hipStream_t stream) {
    (void)in_sizes; (void)n_in; (void)out_size; (void)ws_size;
    const int HEADS[3] = {8, 8, 1};

    auto f = [&](int i) { return (const float*)d_in[i]; };
    const float* X0 = f(0);
    const float* E0 = f(1);
    const int*   A0 = (const int*)d_in[2];

    int id = 3;
    LayerPtrs LP[NLAY];
    for (int i = 0; i < NLAY; ++i) {
        LayerPtrs& P = LP[i];
        P.ln1_g = f(id++); P.ln1_b = f(id++); P.np_w = f(id++); P.np_b = f(id++);
        P.ln2_g = f(id++); P.ln2_b = f(id++);
        P.em_w1 = f(id++); P.em_b1 = f(id++); P.em_w2 = f(id++); P.em_b2 = f(id++);
        P.attn  = f(id++);
        P.op_w  = f(id++); P.op_b = f(id++); P.lno_g = f(id++); P.lno_b = f(id++);
        P.ga_in_w = f(id++); P.ga_in_b = f(id++); P.ga_out_w = f(id++); P.ga_out_b = f(id++);
        P.gamma = f(id++); P.n_g = f(id++); P.n_b = f(id++);
        P.mlp_w1 = f(id++); P.mlp_b1 = f(id++); P.mlp_w2 = f(id++); P.mlp_b2 = f(id++);
        if (i == 0) { P.rp_w = f(id++); P.rp_b = f(id++); }
        else        { P.rp_w = nullptr; P.rp_b = nullptr; }
    }
    const float* pa_w = f(id++); const float* pa_b = f(id++);
    const float* fp_w1 = f(id++); const float* fp_b1 = f(id++);
    const float* fp_ln_g = f(id++); const float* fp_ln_b = f(id++);
    const float* fp_w2 = f(id++); const float* fp_b2 = f(id++);

    // ---------------- workspace layout (floats) ----------------
    float* W = (float*)d_ws;
    size_t o = 0;
    auto alloc = [&](size_t n) { float* p = W + o; o += n; return p; };
    const long MR  = (long)BATCH * NNODE;          // 4096 node rows
    const long EDG = (long)BATCH * NNODE * NNODE;  // 2,097,152 edges

    float* eBuf    = alloc(EDG * EF);              // 128 MB
    float* attnBuf = alloc((size_t)BATCH * 8 * NNODE * NNODE); // 64 MB
    float* logitE  = alloc(EDG);                   // 8 MB
    float* xAb   = alloc(MR * OUTF);
    float* xBb   = alloc(MR * OUTF);
    float* xn    = alloc(MR * OUTF);
    float* hbuf  = alloc(MR * OUTF);
    float* msg   = alloc(MR * OUTF);
    float* aout  = alloc(MR * OUTF);
    float* x2    = alloc(MR * OUTF);
    float* respb = alloc(MR * OUTF);
    float* gout  = alloc(MR * OUTF);
    float* gproj = alloc(MR * OUTF);
    float* fused = alloc(MR * OUTF);
    float* fln   = alloc(MR * OUTF);
    float* m1    = alloc(MR * OUTF);
    float* qkv   = alloc(MR * 3 * OUTF);
    float* li    = alloc(MR * 8);
    float* lj    = alloc(MR * 8);
    float* a_sum = alloc(1024);
    float* pooled = alloc((size_t)BATCH * OUTF);

    const float* xcur = X0;
    for (int i = 0; i < NLAY; ++i) {
        const LayerPtrs& P = LP[i];
        const int inf = (i == 0) ? INF0 : OUTF;
        const int H   = HEADS[i];
        const int hs  = OUTF / H;
        const int hd  = OUTF / H;
        const bool leaky = (i < 2);

        // ---- GAT node path ----
        ln_kernel<<<MR, 256, 0, stream>>>(xcur, P.ln1_g, P.ln1_b, xn, inf, 0);
        gemm_w(stream, xn, inf, P.np_w, inf, P.np_b, nullptr, 0,
               hbuf, OUTF, (int)MR, OUTF, inf, 0);

        // ---- edge MLP (fused, in place after layer 0) ----
        asum_kernel<<<4, 256, 0, stream>>>(P.attn, a_sum, H, 2 * hs + EF);
        edge_kernel<<<(int)(EDG / 256), 256, 0, stream>>>(
            (i == 0) ? E0 : eBuf, eBuf, logitE,
            P.ln2_g, P.ln2_b, P.em_w1, P.em_b1, P.em_w2, P.em_b2, a_sum + 2 * hs);

        // ---- attention logits + softmax ----
        {
            long tot = MR * H;
            logit_kernel<<<(int)((tot + 255) / 256), 256, 0, stream>>>(hbuf, a_sum, li, lj, H, hs);
            long rows = (long)BATCH * H * NNODE;
            coeffs_kernel<<<(int)(rows / 8), 256, 0, stream>>>(li, lj, logitE, A0, attnBuf, H);
        }

        // ---- msg[b,n,h,:] = coeffs[b,h,n,:] @ h[b,:,h,:] (batched over b*h) ----
        gemm(stream,
             attnBuf, NNODE, (long)H * NNODE * NNODE, (long)NNODE * NNODE,
             hbuf, OUTF, (long)NNODE * OUTF, hs, /*kmajor=*/1,
             nullptr, nullptr, 0, 0, 0,
             msg, OUTF, (long)NNODE * OUTF, hs,
             NNODE, hs, NNODE, BATCH * H, H, 1.f, 0);

        // ---- residual projection ----
        const float* resp;
        if (i == 0) {
            gemm_w(stream, xcur, INF0, P.rp_w, INF0, P.rp_b, nullptr, 0,
                   respb, OUTF, (int)MR, OUTF, INF0, 0);
            resp = respb;
        } else {
            resp = xcur;
        }

        // ---- out = lrelu?(msg @ op_w^T + op_b + resp); x2 = LN(out) ----
        gemm_w(stream, msg, OUTF, P.op_w, OUTF, P.op_b, resp, OUTF,
               aout, OUTF, (int)MR, OUTF, OUTF, leaky ? 1 : 0);
        ln_kernel<<<MR, 256, 0, stream>>>(aout, P.lno_g, P.lno_b, x2, OUTF, 0);

        // ---- MHA self-attention on x2 ----
        gemm_w(stream, x2, OUTF, P.ga_in_w, OUTF, P.ga_in_b, nullptr, 0,
               qkv, 3 * OUTF, (int)MR, 3 * OUTF, OUTF, 0);
        // scores = q @ k^T / sqrt(hd)   (batched over b*h)
        gemm(stream,
             qkv, 3 * OUTF, (long)NNODE * 3 * OUTF, hd,
             qkv + OUTF, 3 * OUTF, (long)NNODE * 3 * OUTF, hd, /*kmajor=*/0,
             nullptr, nullptr, 0, 0, 0,
             attnBuf, NNODE, (long)H * NNODE * NNODE, (long)NNODE * NNODE,
             NNODE, NNODE, hd, BATCH * H, H, rsqrtf((float)hd), 0);
        {
            long rows = (long)BATCH * H * NNODE;
            softmax512_kernel<<<(int)(rows / 8), 256, 0, stream>>>(attnBuf);
        }
        // o = attn @ v
        gemm(stream,
             attnBuf, NNODE, (long)H * NNODE * NNODE, (long)NNODE * NNODE,
             qkv + 2 * OUTF, 3 * OUTF, (long)NNODE * 3 * OUTF, hd, /*kmajor=*/1,
             nullptr, nullptr, 0, 0, 0,
             gout, OUTF, (long)NNODE * OUTF, hd,
             NNODE, hd, NNODE, BATCH * H, H, 1.f, 0);
        gemm_w(stream, gout, OUTF, P.ga_out_w, OUTF, P.ga_out_b, nullptr, 0,
               gproj, OUTF, (int)MR, OUTF, OUTF, 0);

        // ---- fuse + norm + MLP ----
        add3_kernel<<<(int)(MR * OUTF / 256), 256, 0, stream>>>(resp, x2, gproj, P.gamma,
                                                                fused, MR * OUTF);
        ln_kernel<<<MR, 256, 0, stream>>>(fused, P.n_g, P.n_b, fln, OUTF, 0);
        gemm_w(stream, fln, OUTF, P.mlp_w1, OUTF, P.mlp_b1, nullptr, 0,
               m1, OUTF, (int)MR, OUTF, OUTF, 3 /*celu*/);
        float* xnext = (i & 1) ? xAb : xBb;
        gemm_w(stream, m1, OUTF, P.mlp_w2, OUTF, P.mlp_b2, nullptr, 0,
               xnext, OUTF, (int)MR, OUTF, OUTF, 0);
        xcur = xnext;
    }

    // ---- pooling ----
    pool1_kernel<<<BATCH, 256, 0, stream>>>(xcur, pa_w, pa_b, pooled);
    pool2_kernel<<<BATCH, POOLH, 0, stream>>>(pooled, fp_w1, fp_b1, fp_ln_g, fp_ln_b,
                                              fp_w2, fp_b2, (float*)d_out);
}
